// ConvolutionalReverb_72206990180763
// MI455X (gfx1250) — matrix operand order repaired and checked by reading, not compiled, not run on hardware
//
#include <hip/hip_runtime.h>
#include <math.h>

typedef __attribute__((ext_vector_type(16))) _Float16 v16h;
typedef __attribute__((ext_vector_type(8)))  _Float16 v8h;
typedef __attribute__((ext_vector_type(8)))  float    v8f;

#define GPL 128        // left guard (elements) in each shifted h copy
#define NBATCH 16      // batches (M dim of WMMA tiles)
#define NG 8           // time tiles per wave (128 output columns/wave)

// Convert x (f32, zero-padded) into an f16 image with PADL leading zeros per row.
__global__ void cr_prep_x(const float* __restrict__ x, _Float16* __restrict__ xf,
                          int T, int XS, int PADL) {
  size_t total  = (size_t)NBATCH * (size_t)XS;
  size_t stride = (size_t)gridDim.x * blockDim.x;
  for (size_t i = (size_t)blockIdx.x * blockDim.x + threadIdx.x; i < total; i += stride) {
    size_t row = i / (size_t)XS;
    int    col = (int)(i - row * (size_t)XS);
    int    c   = col - PADL;
    float  v   = 0.0f;
    if (c >= 0 && c < T) v = x[row * (size_t)T + (size_t)c];
    xf[i] = (_Float16)v;
  }
}

// Build 8 byte-shifted, zero-guarded f16 copies of the effective IR.
// Copy r, element i holds h_logical[p] with p = i - GPL + r, where
// h_logical[p] = (p==K-1) ? 1.0 : tanh(ir_param[p]) for 0<=p<K, else 0.
__global__ void cr_prep_h(const float* __restrict__ irp, _Float16* __restrict__ hc,
                          int K, int SZ) {
  size_t total  = (size_t)8 * (size_t)SZ;
  size_t stride = (size_t)gridDim.x * blockDim.x;
  for (size_t i = (size_t)blockIdx.x * blockDim.x + threadIdx.x; i < total; i += stride) {
    int r   = (int)(i / (size_t)SZ);
    int idx = (int)(i - (size_t)r * (size_t)SZ);
    int p   = idx - GPL + r;
    float v = 0.0f;
    if (p >= 0 && p < K) v = (p == K - 1) ? 1.0f : tanhf(irp[p]);
    hc[i] = (_Float16)v;
  }
}

// Toeplitz-GEMM FIR: one wave computes 16 batches x 128 time columns.
// Per 32-lag chunk: 8x v_wmma_f32_16x16x32_f16 sharing one Toeplitz B fragment.
__global__ __launch_bounds__(128)
void cr_conv_wmma(const _Float16* __restrict__ xf, const _Float16* __restrict__ hc,
                  float* __restrict__ out, int T, int K, int XS, int SZ, int PADL) {
  const int  lane  = threadIdx.x & 31;
  const int  wave  = threadIdx.x >> 5;
  const int  n     = lane & 15;          // N index (time within tile) / A row (batch)
  const bool loH   = lane < 16;
  const int  kbase = loH ? 0 : 8;        // B fragment K base for this half-wave
  const int  khalf = loH ? 0 : 8;        // A fragment column sub-offset

  const long t0w = ((long)blockIdx.x * 4 + wave) * 128;  // first output column
  if (t0w >= (long)T) return;

  // A: row n of padded x image, starting at source column s0_start (+khalf)
  const long s0s = t0w - (long)K - 32;                    // multiple of 32
  const _Float16* aptr = xf + (size_t)n * (size_t)XS
                            + (size_t)(PADL + s0s + khalf);

  // B: per-lane shifted-copy selection; p0 advances by 32 per chunk so
  // r is loop-invariant and the base stays 16B-aligned.
  const int p0u   = 95 - n + kbase;      // (K-1 - C_init - n + kbase) + GPL, C_init = K+32
  const int r     = p0u & 7;
  const int baseu = p0u - r;             // multiple of 8 elements
  const _Float16* bptr = hc + (size_t)r * (size_t)SZ + (size_t)baseu;

  const int NC = (K >> 5) + 2;           // chunks of 32 lags, guard zeros absorb edges

  v8f acc[NG];
#pragma unroll
  for (int g = 0; g < NG; ++g) acc[g] = (v8f){0.f,0.f,0.f,0.f,0.f,0.f,0.f,0.f};

  for (int c = 0; c < NC; ++c) {
    // Toeplitz B fragment: two 8xf16 runs at +0 and +16 from the aligned shifted copy.
    v8h b0 = *(const v8h*)(bptr);
    v8h b1 = *(const v8h*)(bptr + 16);
    v16h b = __builtin_shufflevector(b0, b1,
                                     0,1,2,3,4,5,6,7,8,9,10,11,12,13,14,15);
    // A halves: 9 aligned 8xf16 runs shared between adjacent tiles.
    v8h ah[NG + 1];
#pragma unroll
    for (int j = 0; j <= NG; ++j) ah[j] = *(const v8h*)(aptr + 16 * j);

#pragma unroll
    for (int g = 0; g < NG; ++g) {
      v16h a = __builtin_shufflevector(ah[g], ah[g + 1],
                                       0,1,2,3,4,5,6,7,8,9,10,11,12,13,14,15);
      acc[g] = __builtin_amdgcn_wmma_f32_16x16x32_f16(
          /*neg_a=*/false, a, /*neg_b=*/false, b,
          /*c_mod=*/(short)0, acc[g], /*reuse_a=*/false, /*reuse_b=*/false);
    }
    aptr += 32;
    bptr += 32;
  }

  // C/D layout: VGPR v -> M = v (lanes 0-15) / v+8 (lanes 16-31); N = lane&15.
#pragma unroll
  for (int g = 0; g < NG; ++g) {
    long t = t0w + 16 * g + n;
    if (t < (long)T) {
#pragma unroll
      for (int v = 0; v < 8; ++v) {
        int brow = loH ? v : v + 8;
        out[(size_t)brow * (size_t)T + (size_t)t] = acc[g][v];
      }
    }
  }
}

extern "C" void kernel_launch(void* const* d_in, const int* in_sizes, int n_in,
                              void* d_out, int out_size, void* d_ws, size_t ws_size,
                              hipStream_t stream) {
  const float* x   = (const float*)d_in[0];   // (16,1,960000) f32
  const float* irp = (const float*)d_in[1];   // (1,1,K-1) f32
  float*       out = (float*)d_out;           // (16,1,T) f32

  const int K = in_sizes[1] + 1;              // appended 1.0 tap
  const int T = in_sizes[0] / NBATCH;

  const int PADL = ((K + 64 + 7) / 8) * 8;    // left zero pad >= K+62, 8-aligned
  int XS = PADL + T + 256;                    // padded row length
  XS = ((XS + 7) / 8) * 8;                    // keep rows 16B-aligned
  const int SZ = ((GPL + K + 384 + 7) / 8) * 8; // per-copy length with guards

  char* ws = (char*)d_ws;
  _Float16* xf = (_Float16*)ws;
  size_t xf_bytes = (size_t)NBATCH * (size_t)XS * sizeof(_Float16);
  _Float16* hc = (_Float16*)(ws + ((xf_bytes + 255) & ~(size_t)255));

  // Prep: f16 conversion + padding (fully overwrites its ws regions every call).
  {
    size_t tot = (size_t)NBATCH * (size_t)XS;
    int blocks = (int)((tot + 255) / 256);
    if (blocks > 16384) blocks = 16384;       // grid-stride handles the rest
    cr_prep_x<<<blocks, 256, 0, stream>>>(x, xf, T, XS, PADL);
  }
  {
    size_t tot = (size_t)8 * (size_t)SZ;
    int blocks = (int)((tot + 255) / 256);
    if (blocks > 16384) blocks = 16384;
    cr_prep_h<<<blocks, 256, 0, stream>>>(irp, hc, K, SZ);
  }

  // Main Toeplitz-WMMA convolution: 1 wave per 128 output columns, 4 waves/block.
  int waves  = (T + 127) / 128;               // 7500 for T=960000
  int blocks = (waves + 3) / 4;               // 1875
  cr_conv_wmma<<<blocks, 128, 0, stream>>>(xf, hc, out, T, K, XS, SZ, PADL);
}
